// GraphNeuralNet_27839978012819
// MI455X (gfx1250) — compile-verified, matrix operand order checked by reading, not run on hardware
//
#include <hip/hip_runtime.h>

// ---------------- constants ----------------
constexpr int NN   = 50000;   // nodes
constexpr int NE   = 800000;  // edges
constexpr int NG   = 256;     // graphs
constexpr int D0   = 336;     // input feat
constexpr int DH   = 100;     // hidden feat
constexpr int NPAD = 112;     // hidden padded to 7*16 (output cols)
constexpr int K0P  = 352;     // 336 padded to 11*32
constexpr int KHP  = 128;     // 100 padded to 4*32
constexpr int HBW  = 128;     // bf16 activation row stride (K-padded)

typedef __attribute__((ext_vector_type(8)))  __bf16 v8bf;
typedef __attribute__((ext_vector_type(16))) __bf16 v16bf;
typedef __attribute__((ext_vector_type(8)))  float  v8f;

__device__ inline unsigned short f32_to_bf16(float f) {
    unsigned u = __float_as_uint(f);
    u += 0x7FFFu + ((u >> 16) & 1u);   // round-to-nearest-even
    return (unsigned short)(u >> 16);
}

// ---------------- staging kernels ----------------
__global__ void __launch_bounds__(256) convert_x_kernel(const float* __restrict__ x,
                                                        unsigned short* __restrict__ xb) {
    size_t t = (size_t)blockIdx.x * 256 + threadIdx.x;
    if (t >= (size_t)NN * K0P) return;
    size_t r = t / K0P;
    int    c = (int)(t - r * K0P);
    float  v = (c < D0) ? x[r * D0 + c] : 0.0f;
    xb[t] = f32_to_bf16(v);
}

__global__ void __launch_bounds__(256) pad_weights_kernel(const float* __restrict__ w,
                                                          unsigned short* __restrict__ dst,
                                                          int rows, int cols, int Kp) {
    int t = blockIdx.x * 256 + threadIdx.x;
    if (t >= NPAD * Kp) return;
    int r = t / Kp, c = t - r * Kp;
    float v = (r < rows && c < cols) ? w[r * cols + c] : 0.0f;
    dst[t] = f32_to_bf16(v);
}

// ---------------- dual GEMM: outRel = A@Wrel^T, outBase = A@Wroot^T + bias ----------------
// A: [NN, Kp] bf16 (zero K-padded). Wrel/Wroot: [NPAD, Kp] bf16 (zero padded).
// One wave -> one 16-row tile, ALL 7 column tiles, BOTH weights: a single A-tile
// load feeds 14 WMMAs per K-step (A re-read eliminated; weights are L0-resident).
template <int Kp>
__global__ void __launch_bounds__(256) gemm_dual_wmma(const unsigned short* __restrict__ A,
                                                      const unsigned short* __restrict__ Wrel,
                                                      const unsigned short* __restrict__ Wroot,
                                                      const float* __restrict__ bias,
                                                      float* __restrict__ outRel,
                                                      float* __restrict__ outBase) {
    constexpr int CT = NPAD / 16;   // 7 column tiles
    constexpr int KT = Kp / 32;
    const int rowTiles = NN / 16;
    int wave = blockIdx.x * (blockDim.x >> 5) + (threadIdx.x >> 5);
    if (wave >= rowTiles) return;   // wave-uniform: EXEC stays all-1s for WMMA
    int lane = threadIdx.x & 31;
    int half = lane >> 4, lr = lane & 15;

    const unsigned short* Ap = A    + (size_t)(wave * 16 + lr) * Kp;
    const unsigned short* Wr = Wrel + (size_t)lr * Kp;   // + ct*16*Kp is a compile-time offset
    const unsigned short* Wo = Wroot + (size_t)lr * Kp;

    v8f accR[CT] = {};
    v8f accO[CT] = {};

#pragma unroll
    for (int kt = 0; kt < KT; ++kt) {
        const int kb = (kt << 5) + (half << 3);   // ISA lane layout: K {kb..kb+7, kb+16..kb+23}
        v8bf a0 = *(const v8bf*)(Ap + kb);
        v8bf a1 = *(const v8bf*)(Ap + kb + 16);
        v16bf av = __builtin_shufflevector(a0, a1, 0,1,2,3,4,5,6,7,8,9,10,11,12,13,14,15);
#pragma unroll
        for (int ct = 0; ct < CT; ++ct) {
            v8bf r0 = *(const v8bf*)(Wr + ct * 16 * Kp + kb);
            v8bf r1 = *(const v8bf*)(Wr + ct * 16 * Kp + kb + 16);
            v8bf o0 = *(const v8bf*)(Wo + ct * 16 * Kp + kb);
            v8bf o1 = *(const v8bf*)(Wo + ct * 16 * Kp + kb + 16);
            v16bf rv = __builtin_shufflevector(r0, r1, 0,1,2,3,4,5,6,7,8,9,10,11,12,13,14,15);
            v16bf ov = __builtin_shufflevector(o0, o1, 0,1,2,3,4,5,6,7,8,9,10,11,12,13,14,15);
            accR[ct] = __builtin_amdgcn_wmma_f32_16x16x32_bf16(false, av, false, rv, (short)0, accR[ct], false, false);
            accO[ct] = __builtin_amdgcn_wmma_f32_16x16x32_bf16(false, av, false, ov, (short)0, accO[ct], false, false);
        }
    }

    const int rowBase = wave * 16 + half * 8;   // C/D layout: vgpr r -> row r + 8*(lane/16)
#pragma unroll
    for (int ct = 0; ct < CT; ++ct) {
        int col = ct * 16 + lr;
        float bv = (col < DH) ? bias[col] : 0.0f;   // padded cols stay exactly 0
#pragma unroll
        for (int r = 0; r < 8; ++r) {
            size_t idx = (size_t)(rowBase + r) * NPAD + col;
            outRel[idx]  = accR[ct][r];
            outBase[idx] = accO[ct][r] + bv;
        }
    }
}

// ---------------- edge scatter: base[dst] += xr[src] ----------------
__global__ void __launch_bounds__(256) scatter_add_kernel(const int* __restrict__ ei,
                                                          const float* __restrict__ xr,
                                                          float* __restrict__ basep) {
    int wave = blockIdx.x * (blockDim.x >> 5) + (threadIdx.x >> 5);
    if (wave >= NE) return;
    int lane = threadIdx.x & 31;
    int s = ei[wave];
    int d = ei[NE + wave];
    const float* sp = xr    + (size_t)s * NPAD;
    float*       dp = basep + (size_t)d * NPAD;
    for (int k = lane; k < DH; k += 32) atomicAdd(dp + k, sp[k]);
}

// ---------------- relu (in-place fp32) + bf16 activation for next GEMM ----------------
__global__ void __launch_bounds__(256) relu_convert_kernel(float* __restrict__ basep,
                                                           unsigned short* __restrict__ hbf) {
    size_t t = (size_t)blockIdx.x * 256 + threadIdx.x;
    if (t >= (size_t)NN * HBW) return;
    int    c = (int)(t & (HBW - 1));
    size_t r = t >> 7;
    float v = 0.0f;
    if (c < NPAD) {
        float* p = basep + r * NPAD + c;
        v = fmaxf(*p, 0.0f);
        *p = v;
    }
    hbf[t] = f32_to_bf16(v);
}

// ---------------- pooling ----------------
__global__ void __launch_bounds__(256) zero_pool_kernel(float* __restrict__ sums,
                                                        float* __restrict__ counts) {
    int t = blockIdx.x * 256 + threadIdx.x;
    if (t < NG * DH) sums[t] = 0.0f;
    if (t < NG) counts[t] = 0.0f;
}

__global__ void __launch_bounds__(256) pool_kernel(const float* __restrict__ basep,
                                                   const int* __restrict__ batch,
                                                   float* __restrict__ sums,
                                                   float* __restrict__ counts) {
    size_t t = (size_t)blockIdx.x * 256 + threadIdx.x;
    if (t >= (size_t)NN * DH) return;
    int node = (int)(t / DH);
    int d    = (int)(t - (size_t)node * DH);
    int g    = batch[node];
    atomicAdd(&sums[(size_t)g * DH + d], basep[(size_t)node * NPAD + d]);
    if (d == 0) atomicAdd(&counts[g], 1.0f);
}

// ---------------- tiny MLP head (one block per graph) ----------------
__global__ void __launch_bounds__(128) mlp_kernel(const float* __restrict__ sums,
                                                  const float* __restrict__ counts,
                                                  const float* __restrict__ lw1, const float* __restrict__ lb1,
                                                  const float* __restrict__ lw2, const float* __restrict__ lb2,
                                                  const float* __restrict__ lw3, const float* __restrict__ lb3,
                                                  float* __restrict__ out) {
    __shared__ float bufA[DH];
    __shared__ float bufB[DH];
    int g = blockIdx.x, t = threadIdx.x;
    if (t < DH) bufA[t] = sums[(size_t)g * DH + t] / fmaxf(counts[g], 1.0f);
    __syncthreads();
    if (t < DH) {
        float acc = lb1[t];
        for (int k = 0; k < DH; ++k) acc += bufA[k] * lw1[t * DH + k];
        bufB[t] = fmaxf(acc, 0.0f);
    }
    __syncthreads();
    if (t < DH) {
        float acc = lb2[t];
        for (int k = 0; k < DH; ++k) acc += bufB[k] * lw2[t * DH + k];
        bufA[t] = fmaxf(acc, 0.0f);
    }
    __syncthreads();
    if (t < 29) {
        float acc = lb3[t];
        for (int k = 0; k < DH; ++k) acc += bufA[k] * lw3[t * DH + k];
        out[(size_t)g * 29 + t] = acc;
    }
}

// ---------------- host ----------------
extern "C" void kernel_launch(void* const* d_in, const int* in_sizes, int n_in,
                              void* d_out, int out_size, void* d_ws, size_t ws_size,
                              hipStream_t stream) {
    const float* x       = (const float*)d_in[0];
    const int*   ei      = (const int*)d_in[1];
    const int*   batch   = (const int*)d_in[2];
    const float* w1_rel  = (const float*)d_in[3];
    const float* w1_root = (const float*)d_in[4];
    const float* b1      = (const float*)d_in[5];
    const float* w_rel   = (const float*)d_in[6];   // [4,100,100]
    const float* w_root  = (const float*)d_in[7];   // [4,100,100]
    const float* b       = (const float*)d_in[8];   // [4,100]
    const float* lw1     = (const float*)d_in[9];
    const float* lb1     = (const float*)d_in[10];
    const float* lw2     = (const float*)d_in[11];
    const float* lb2     = (const float*)d_in[12];
    const float* lw3     = (const float*)d_in[13];
    const float* lb3     = (const float*)d_in[14];
    float* out = (float*)d_out;

    // scratch carving (256B aligned)
    char* ws = (char*)d_ws;
    size_t off = 0;
    auto take = [&](size_t bytes) -> char* {
        char* p = ws + off;
        off = (off + bytes + 255) & ~(size_t)255;
        return p;
    };
    unsigned short* xb      = (unsigned short*)take((size_t)NN * K0P * 2);   // bf16 x, K-padded
    unsigned short* hbf     = (unsigned short*)take((size_t)NN * HBW * 2);   // bf16 activations
    float*          basep   = (float*)take((size_t)NN * NPAD * 4);           // root part + agg
    float*          xrp     = (float*)take((size_t)NN * NPAD * 4);           // rel-transformed
    unsigned short* w1rel_p  = (unsigned short*)take((size_t)NPAD * K0P * 2);
    unsigned short* w1root_p = (unsigned short*)take((size_t)NPAD * K0P * 2);
    unsigned short* wrel_p   = (unsigned short*)take((size_t)4 * NPAD * KHP * 2);
    unsigned short* wroot_p  = (unsigned short*)take((size_t)4 * NPAD * KHP * 2);
    float*          sums    = (float*)take((size_t)NG * DH * 4);
    float*          counts  = (float*)take((size_t)NG * 4);
    (void)in_sizes; (void)n_in; (void)out_size; (void)ws_size;

    auto blocks = [](size_t n) { return (unsigned)((n + 255) / 256); };

    // stage inputs to bf16 (zero-padded)
    convert_x_kernel<<<blocks((size_t)NN * K0P), 256, 0, stream>>>(x, xb);
    pad_weights_kernel<<<blocks((size_t)NPAD * K0P), 256, 0, stream>>>(w1_rel,  w1rel_p,  DH, D0, K0P);
    pad_weights_kernel<<<blocks((size_t)NPAD * K0P), 256, 0, stream>>>(w1_root, w1root_p, DH, D0, K0P);
    for (int i = 0; i < 4; ++i) {
        pad_weights_kernel<<<blocks((size_t)NPAD * KHP), 256, 0, stream>>>(
            w_rel  + (size_t)i * DH * DH, wrel_p  + (size_t)i * NPAD * KHP, DH, DH, KHP);
        pad_weights_kernel<<<blocks((size_t)NPAD * KHP), 256, 0, stream>>>(
            w_root + (size_t)i * DH * DH, wroot_p + (size_t)i * NPAD * KHP, DH, DH, KHP);
    }

    const int gemmWaves  = NN / 16;                          // one wave per 16-row tile
    const int gemmBlocks = (gemmWaves + 7) / 8;              // 8 waves / 256-thread block
    const int scatBlocks = NE / 8;                           // 1 wave per edge

    // layer 1 (336 -> 100): transform-first, then scatter, then relu
    gemm_dual_wmma<K0P><<<gemmBlocks, 256, 0, stream>>>(xb, w1rel_p, w1root_p, b1, xrp, basep);
    scatter_add_kernel<<<scatBlocks, 256, 0, stream>>>(ei, xrp, basep);
    relu_convert_kernel<<<blocks((size_t)NN * HBW), 256, 0, stream>>>(basep, hbf);

    // layers 2..5 (100 -> 100)
    for (int i = 0; i < 4; ++i) {
        gemm_dual_wmma<KHP><<<gemmBlocks, 256, 0, stream>>>(
            hbf, wrel_p + (size_t)i * NPAD * KHP, wroot_p + (size_t)i * NPAD * KHP,
            b + (size_t)i * DH, xrp, basep);
        scatter_add_kernel<<<scatBlocks, 256, 0, stream>>>(ei, xrp, basep);
        relu_convert_kernel<<<blocks((size_t)NN * HBW), 256, 0, stream>>>(basep, hbf);
    }

    // global mean pool + MLP head
    zero_pool_kernel<<<blocks((size_t)NG * DH), 256, 0, stream>>>(sums, counts);
    pool_kernel<<<blocks((size_t)NN * DH), 256, 0, stream>>>(basep, batch, sums, counts);
    mlp_kernel<<<NG, 128, 0, stream>>>(sums, counts, lw1, lb1, lw2, lb2, lw3, lb3, out);
}
